// TAD_GAT_10943576670617
// MI455X (gfx1250) — matrix operand order repaired and checked by reading, not compiled, not run on hardware
//
#include <hip/hip_runtime.h>
#include <hip/hip_bf16.h>
#include <cstring>

#define HEADS   4
#define HID     64
#define TSTEPS  3
#define NFD     57
#define RTD     78
#define BATCH   4
#define NNODES  10000
#define BN_TOTAL (BATCH*NNODES)
#define CCH     (HEADS*HID)      /* 256 */
#define XFEAT   (NFD + TSTEPS*RTD) /* 291 */
#define SLOPE   0.2f
#define EPSBN   1e-5f

typedef __bf16 bf16_t;
typedef bf16_t v16bf __attribute__((ext_vector_type(16)));
typedef float  v8f   __attribute__((ext_vector_type(8)));

// ---------------------------------------------------------------- helpers
__device__ __forceinline__ unsigned short f2bf(float f) {
  unsigned int u = __float_as_uint(f);
  u = (u + 0x7FFFu + ((u >> 16) & 1u)) >> 16;   // RNE fp32 -> bf16
  return (unsigned short)u;
}
__device__ __forceinline__ float sigm(float x) { return 1.f / (1.f + __expf(-x)); }

__device__ __forceinline__ v8f wmma_bf16(v16bf a, v16bf b, v8f c) {
  return __builtin_amdgcn_wmma_f32_16x16x32_bf16(false, a, false, b, (short)0, c,
                                                 false, false);
}

// A fragment 16x32 (M x K), ISA 7.12.2 16-bit A layout:
//   row = lane&15 ; element e -> K = (e&7) + 16*(e>>3) + 8*(lane>>4)
__device__ __forceinline__ v16bf load_a_frag(const unsigned short* abuf, int ld,
                                             int row, int kbase, int half) {
  union { v16bf v; unsigned short s[16]; } u;
#pragma unroll
  for (int e = 0; e < 16; ++e) {
    int k = kbase + (e & 7) + ((e >> 3) << 4) + (half << 3);
    u.s[e] = abuf[row * ld + k];
  }
  return u.v;
}

// B fragment 32x16 (K x N), pre-packed in workspace: index = ((kt*NT+nt)*32+lane)*16+e
__device__ __forceinline__ v16bf load_b_frag(const unsigned short* __restrict__ wp,
                                             int kt, int nt, int NT, int lane) {
  const uint4* p = (const uint4*)(wp + ((((kt * NT) + nt) * 32 + lane) << 4));
  union { uint4 q[2]; v16bf v; } u;
  u.q[0] = p[0]; u.q[1] = p[1];
  return u.v;
}

// monotone float <-> ordered-uint encoding for atomicMax on floats
__device__ __forceinline__ unsigned int enc_f(float f) {
  unsigned int b = __float_as_uint(f);
  return (b & 0x80000000u) ? ~b : (b | 0x80000000u);
}

// ---------------------------------------------------------------- fills
__global__ void fill_f32_kernel(float* p, float v, int n) {
  int i = blockIdx.x * blockDim.x + threadIdx.x;
  if (i < n) p[i] = v;
}
__global__ void fill_u32_kernel(unsigned int* p, unsigned int v, int n) {
  int i = blockIdx.x * blockDim.x + threadIdx.x;
  if (i < n) p[i] = v;
}

// ---------------------------------------------------------------- weight pack
// W is [Nreal][Kreal] row-major (torch-style, y = x @ W^T). B[k][n] = W[n][k].
// Packed per ISA B layout: element e of lane-half covers K = kt*32 + (e&7) + 16*(e>>3) + 8*half.
__global__ void pack_weight_kernel(const float* __restrict__ W, unsigned short* out,
                                   int Kreal, int KT, int Nreal, int NT) {
  int idx = blockIdx.x * blockDim.x + threadIdx.x;
  int total = KT * NT * 512;
  if (idx >= total) return;
  int e    = idx & 15;
  int lane = (idx >> 4) & 31;
  int fn   = idx >> 9;
  int nt   = fn % NT, kt = fn / NT;
  int half = lane >> 4;
  int c    = nt * 16 + (lane & 15);
  int k    = kt * 32 + (e & 7) + 16 * (e >> 3) + 8 * half;
  float v  = (k < Kreal && c < Nreal) ? W[c * Kreal + k] : 0.f;
  out[idx] = f2bf(v);
}

// ---------------------------------------------------------------- fused LSTM +
// temporal attention + static branch.  64 threads = 2 waves; each wave = 16 nodes.
__global__ void __launch_bounds__(64) lstm_fused_kernel(
    const float* __restrict__ x, const unsigned short* __restrict__ wih_p,
    const unsigned short* __restrict__ whh_p, const unsigned short* __restrict__ wn_p,
    const float* __restrict__ bih, const float* __restrict__ bhh,
    const float* __restrict__ Wa, const float* __restrict__ ba,
    const float* __restrict__ bnb, float* __restrict__ h0out) {
  __shared__ unsigned short sA[2][16][96];          // staged A tiles (bf16)
  __shared__ unsigned short sH[2][2][16][64];       // h ping-pong (bf16)
  __shared__ float          sSeq[2][TSTEPS][16][64];// h_t history (fp32)

  const int tid  = threadIdx.x;
  const int wv   = tid >> 5;
  const int lane = tid & 31;
  const int half = lane >> 4;
  const int col  = lane & 15;
  const int bn0  = (blockIdx.x * 2 + wv) * 16;

  float cst[8][4];                                  // c-state: rows j+8*half, u-group
#pragma unroll
  for (int j = 0; j < 8; ++j)
#pragma unroll
    for (int u = 0; u < 4; ++u) cst[j][u] = 0.f;

  for (int t = 0; t < TSTEPS; ++t) {
    // stage x_t (16 x 78, padded to 96) as bf16
    for (int idx = lane; idx < 16 * 96; idx += 32) {
      int r = idx / 96, k = idx - r * 96;
      float v = (k < RTD) ? x[(size_t)(bn0 + r) * XFEAT + NFD + t * RTD + k] : 0.f;
      sA[wv][r][k] = f2bf(v);
    }
    __syncthreads();
    const int pr = t & 1, pw = pr ^ 1;
#pragma unroll
    for (int ut = 0; ut < 4; ++ut) {
      v8f ai = {}, af = {}, ag = {}, ao = {};
#pragma unroll
      for (int kt = 0; kt < 3; ++kt) {              // K = 78 -> 3 tiles of 32
        v16bf a = load_a_frag(&sA[wv][0][0], 96, col, kt * 32, half);
        ai = wmma_bf16(a, load_b_frag(wih_p, kt, ut,      16, lane), ai);
        af = wmma_bf16(a, load_b_frag(wih_p, kt, 4 + ut,  16, lane), af);
        ag = wmma_bf16(a, load_b_frag(wih_p, kt, 8 + ut,  16, lane), ag);
        ao = wmma_bf16(a, load_b_frag(wih_p, kt, 12 + ut, 16, lane), ao);
      }
      if (t > 0) {                                  // recurrent h @ Whh^T (K = 64)
#pragma unroll
        for (int kt = 0; kt < 2; ++kt) {
          v16bf a = load_a_frag(&sH[wv][pr][0][0], 64, col, kt * 32, half);
          ai = wmma_bf16(a, load_b_frag(whh_p, kt, ut,      16, lane), ai);
          af = wmma_bf16(a, load_b_frag(whh_p, kt, 4 + ut,  16, lane), af);
          ag = wmma_bf16(a, load_b_frag(whh_p, kt, 8 + ut,  16, lane), ag);
          ao = wmma_bf16(a, load_b_frag(whh_p, kt, 12 + ut, 16, lane), ao);
        }
      }
      const int   u   = ut * 16 + col;
      const float bi  = bih[u]        + bhh[u];
      const float bfv = bih[64 + u]   + bhh[64 + u];
      const float bg  = bih[128 + u]  + bhh[128 + u];
      const float bo  = bih[192 + u]  + bhh[192 + u];
#pragma unroll
      for (int j = 0; j < 8; ++j) {
        int   row = j + 8 * half;
        float iv = sigm(ai[j] + bi);
        float fv = sigm(af[j] + bfv);
        float gv = tanhf(ag[j] + bg);
        float ov = sigm(ao[j] + bo);
        float cc = fv * cst[j][ut] + iv * gv;
        float hh = ov * tanhf(cc);
        cst[j][ut] = cc;
        sSeq[wv][t][row][u] = hh;
        sH[wv][pw][row][u]  = f2bf(hh);
      }
    }
    __syncthreads();
  }

  // temporal softmax-attention: lane r (<16) owns row r
  if (lane < 16) {
    int   row = lane;
    float l0 = ba[0], l1 = ba[0], l2 = ba[0];
    for (int u = 0; u < HID; ++u) {
      float w = Wa[u];
      l0 += sSeq[wv][0][row][u] * w;
      l1 += sSeq[wv][1][row][u] * w;
      l2 += sSeq[wv][2][row][u] * w;
    }
    float m  = fmaxf(l0, fmaxf(l1, l2));
    float w0 = __expf(l0 - m), w1 = __expf(l1 - m), w2 = __expf(l2 - m);
    float inv = 1.f / (w0 + w1 + w2);
    w0 *= inv; w1 *= inv; w2 *= inv;
    for (int u = 0; u < HID; ++u) {
      float v = w0 * sSeq[wv][0][row][u] + w1 * sSeq[wv][1][row][u] +
                w2 * sSeq[wv][2][row][u];
      sSeq[wv][0][row][u] = v;                      // h_temp
    }
  }
  __syncthreads();

  // static branch: relu(node_f @ Wn^T + bnb), K = 57 padded to 64
  for (int idx = lane; idx < 16 * 64; idx += 32) {
    int r = idx >> 6, k = idx & 63;
    float v = (k < NFD) ? x[(size_t)(bn0 + r) * XFEAT + k] : 0.f;
    sA[wv][r][k] = f2bf(v);
  }
  __syncthreads();
#pragma unroll
  for (int nt = 0; nt < 4; ++nt) {
    v8f acc = {};
#pragma unroll
    for (int kt = 0; kt < 2; ++kt) {
      v16bf a = load_a_frag(&sA[wv][0][0], 96, col, kt * 32, half);
      acc = wmma_bf16(a, load_b_frag(wn_p, kt, nt, 4, lane), acc);
    }
    const int   u  = nt * 16 + col;
    const float bb = bnb[u];
#pragma unroll
    for (int j = 0; j < 8; ++j) {
      int row = j + 8 * half;
      float v = fmaxf(acc[j] + bb, 0.f) + sSeq[wv][0][row][u];
      h0out[(size_t)(bn0 + row) * HID + u] = v;
    }
  }
}

// ---------------------------------------------------------------- GAT GEMM:
// xw[bn][256] = h[bn][K] @ W^T.  2 waves/block, 16 rows/wave.
template <int KDIM>
__global__ void __launch_bounds__(64) gemm_xw_kernel(
    const float* __restrict__ hin, const unsigned short* __restrict__ wp,
    float* __restrict__ xw) {
  __shared__ unsigned short sA[2][16][KDIM];
  const int tid = threadIdx.x, wv = tid >> 5, lane = tid & 31;
  const int half = lane >> 4, col = lane & 15;
  const int bn0 = (blockIdx.x * 2 + wv) * 16;
  for (int idx = lane; idx < 16 * KDIM; idx += 32) {
    int r = idx / KDIM, k = idx - r * KDIM;
    sA[wv][r][k] = f2bf(hin[(size_t)(bn0 + r) * KDIM + k]);
  }
  __syncthreads();
#pragma unroll
  for (int nt = 0; nt < 16; ++nt) {
    v8f acc = {};
#pragma unroll
    for (int kt = 0; kt < KDIM / 32; ++kt) {
      v16bf a = load_a_frag(&sA[wv][0][0], KDIM, col, kt * 32, half);
      acc = wmma_bf16(a, load_b_frag(wp, kt, nt, 16, lane), acc);
    }
#pragma unroll
    for (int j = 0; j < 8; ++j) {
      int row = j + 8 * half;
      xw[(size_t)(bn0 + row) * CCH + nt * 16 + col] = acc[j];
    }
  }
}

// ---------------------------------------------------------------- attention coefs
__global__ void attn_coef_kernel(const float* __restrict__ xw,
                                 const float* __restrict__ a_s,
                                 const float* __restrict__ a_d,
                                 float* __restrict__ asrc, float* __restrict__ adst,
                                 int total) {
  int idx = blockIdx.x * blockDim.x + threadIdx.x;
  if (idx >= total) return;
  int bn = idx >> 2, hd = idx & 3;
  const float* row = xw + (size_t)bn * CCH + hd * HID;
  float s = 0.f, d = 0.f;
  for (int c = 0; c < HID; ++c) {
    float v = row[c];
    s += v * a_s[hd * HID + c];
    d += v * a_d[hd * HID + c];
  }
  asrc[idx] = s;
  adst[idx] = d;
}

// ---------------------------------------------------------------- edge phases
__global__ void edge_max_kernel(const int* __restrict__ ei, int E, int EE,
                                const float* __restrict__ asrc,
                                const float* __restrict__ adst,
                                unsigned int* __restrict__ emax_u, int total) {
  int idx = blockIdx.x * blockDim.x + threadIdx.x;
  if (idx >= total) return;
  int b   = idx / (EE * HEADS);
  int rem = idx - b * EE * HEADS;
  int e   = rem >> 2, hd = rem & 3;
  int src = (e < E) ? ei[e]     : (e - E);
  int dst = (e < E) ? ei[E + e] : (e - E);
  float v = asrc[(b * NNODES + src) * HEADS + hd] +
            adst[(b * NNODES + dst) * HEADS + hd];
  v = (v > 0.f) ? v : SLOPE * v;
  atomicMax(&emax_u[(b * NNODES + dst) * HEADS + hd], enc_f(v));
}

__global__ void dec_max_kernel(unsigned int* buf, int n) {
  int i = blockIdx.x * blockDim.x + threadIdx.x;
  if (i >= n) return;
  unsigned int u = buf[i];
  buf[i] = (u & 0x80000000u) ? (u ^ 0x80000000u) : ~u;   // back to float bits
}

__global__ void edge_sum_kernel(const int* __restrict__ ei, int E, int EE,
                                const float* __restrict__ asrc,
                                const float* __restrict__ adst,
                                const float* __restrict__ emaxf,
                                float* __restrict__ den, int total) {
  int idx = blockIdx.x * blockDim.x + threadIdx.x;
  if (idx >= total) return;
  int b   = idx / (EE * HEADS);
  int rem = idx - b * EE * HEADS;
  int e   = rem >> 2, hd = rem & 3;
  int src = (e < E) ? ei[e]     : (e - E);
  int dst = (e < E) ? ei[E + e] : (e - E);
  int di  = (b * NNODES + dst) * HEADS + hd;
  float v = asrc[(b * NNODES + src) * HEADS + hd] + adst[di];
  v = (v > 0.f) ? v : SLOPE * v;
  atomicAdd(&den[di], __expf(v - emaxf[di]));
}

// one wave per (batch, edge); lane handles 8 channels of one head
__global__ void edge_agg_kernel(const int* __restrict__ ei, int E, int EE,
                                const float* __restrict__ asrc,
                                const float* __restrict__ adst,
                                const float* __restrict__ emaxf,
                                const float* __restrict__ den,
                                const float* __restrict__ xw,
                                float* __restrict__ agg, int totalWaves) {
  int gt  = blockIdx.x * blockDim.x + threadIdx.x;
  int wid = gt >> 5, lane = gt & 31;
  if (wid >= totalWaves) return;
  int b   = wid / EE;
  int e   = wid - b * EE;
  int src = (e < E) ? ei[e]     : (e - E);
  int dst = (e < E) ? ei[E + e] : (e - E);
  int hd  = lane >> 3;
  int di  = (b * NNODES + dst) * HEADS + hd;
  float v = asrc[(b * NNODES + src) * HEADS + hd] + adst[di];
  v = (v > 0.f) ? v : SLOPE * v;
  float alpha = __expf(v - emaxf[di]) / (den[di] + 1e-16f);
  const float* srow = xw  + (size_t)(b * NNODES + src) * CCH + lane * 8;
  float*       drow = agg + (size_t)(b * NNODES + dst) * CCH + lane * 8;
#pragma unroll
  for (int k = 0; k < 8; ++k) atomicAdd(&drow[k], alpha * srow[k]);
}

// ---------------------------------------------------------------- bias+relu+BN stats
__global__ void __launch_bounds__(256) bias_relu_stats_kernel(
    float* __restrict__ agg, const float* __restrict__ bias,
    float* __restrict__ sum, float* __restrict__ sq) {
  int   c  = threadIdx.x;
  int   r0 = blockIdx.x * 64;
  float s = 0.f, s2 = 0.f;
  float bb = bias[c];
  for (int r = 0; r < 64; ++r) {
    int row = r0 + r;
    if (row >= BN_TOTAL) break;
    size_t i = (size_t)row * CCH + c;
    float  v = fmaxf(agg[i] + bb, 0.f);
    agg[i] = v;
    s += v;
    s2 += v * v;
  }
  atomicAdd(&sum[c], s);
  atomicAdd(&sq[c], s2);
}

__global__ void bn_apply_kernel(const float* __restrict__ agg,
                                const float* __restrict__ sum,
                                const float* __restrict__ sq,
                                const float* __restrict__ g,
                                const float* __restrict__ be,
                                float* __restrict__ out, int total) {
  int i = blockIdx.x * blockDim.x + threadIdx.x;
  if (i >= total) return;
  int   c   = i & (CCH - 1);
  float mu  = sum[c] * (1.f / BN_TOTAL);
  float var = sq[c] * (1.f / BN_TOTAL) - mu * mu;
  out[i] = (agg[i] - mu) * rsqrtf(var + EPSBN) * g[c] + be[c];
}

// ---------------------------------------------------------------- final projection
__global__ void __launch_bounds__(256) final_dot_kernel(
    const float* __restrict__ h, const float* __restrict__ Wc,
    const float* __restrict__ bc, float* __restrict__ out) {
  int wid  = (blockIdx.x * blockDim.x + threadIdx.x) >> 5;
  int lane = threadIdx.x & 31;
  if (wid >= BN_TOTAL) return;
  const float* row = h + (size_t)wid * CCH;
  float s = 0.f;
  for (int k = lane; k < CCH; k += 32) s += row[k] * Wc[k];
#pragma unroll
  for (int off = 16; off > 0; off >>= 1) s += __shfl_xor(s, off, 32);
  if (lane == 0) out[wid] = s + bc[0];
}

// ---------------------------------------------------------------- host launcher
extern "C" void kernel_launch(void* const* d_in, const int* in_sizes, int n_in,
                              void* d_out, int out_size, void* d_ws, size_t ws_size,
                              hipStream_t stream) {
  (void)n_in; (void)out_size; (void)ws_size;
  const float* x    = (const float*)d_in[0];
  const int*   ei   = (const int*)d_in[1];
  const float* Wih  = (const float*)d_in[2];
  const float* Whh  = (const float*)d_in[3];
  const float* bih  = (const float*)d_in[4];
  const float* bhh  = (const float*)d_in[5];
  const float* Wa   = (const float*)d_in[6];
  const float* ba   = (const float*)d_in[7];
  const float* Wn   = (const float*)d_in[8];
  const float* bnb  = (const float*)d_in[9];
  const float* W1   = (const float*)d_in[10];
  const float* as1  = (const float*)d_in[11];
  const float* ad1  = (const float*)d_in[12];
  const float* b1   = (const float*)d_in[13];
  const float* W2   = (const float*)d_in[14];
  const float* as2  = (const float*)d_in[15];
  const float* ad2  = (const float*)d_in[16];
  const float* b2   = (const float*)d_in[17];
  const float* g1   = (const float*)d_in[18];
  const float* be1  = (const float*)d_in[19];
  const float* g2   = (const float*)d_in[20];
  const float* be2  = (const float*)d_in[21];
  const float* Wc   = (const float*)d_in[22];
  const float* bc   = (const float*)d_in[23];
  float*       out  = (float*)d_out;

  const int E  = in_sizes[1] / 2;
  const int EE = E + NNODES;

  // workspace bump allocator
  char*  base = (char*)d_ws;
  size_t off  = 0;
  auto alloc = [&](size_t bytes) -> void* {
    void* p = base + off;
    off += (bytes + 255) & ~(size_t)255;
    return p;
  };
  unsigned short* wih_p = (unsigned short*)alloc((size_t)3 * 16 * 512 * 2);
  unsigned short* whh_p = (unsigned short*)alloc((size_t)2 * 16 * 512 * 2);
  unsigned short* wn_p  = (unsigned short*)alloc((size_t)2 * 4 * 512 * 2);
  unsigned short* w1_p  = (unsigned short*)alloc((size_t)2 * 16 * 512 * 2);
  unsigned short* w2_p  = (unsigned short*)alloc((size_t)8 * 16 * 512 * 2);
  float* h0   = (float*)alloc((size_t)BN_TOTAL * HID * 4);
  float* xw   = (float*)alloc((size_t)BN_TOTAL * CCH * 4);
  float* agg  = (float*)alloc((size_t)BN_TOTAL * CCH * 4);
  float* hbn  = (float*)alloc((size_t)BN_TOTAL * CCH * 4);
  float* asrc = (float*)alloc((size_t)BN_TOTAL * HEADS * 4);
  float* adst = (float*)alloc((size_t)BN_TOTAL * HEADS * 4);
  unsigned int* emax = (unsigned int*)alloc((size_t)BN_TOTAL * HEADS * 4);
  float* den  = (float*)alloc((size_t)BN_TOTAL * HEADS * 4);
  float* stat = (float*)alloc((size_t)2 * CCH * 4);   // [sum | sumsq]

  auto cdiv = [](int a, int b) { return (a + b - 1) / b; };

  // ---- pack weights (bf16, fragment layout)
  pack_weight_kernel<<<cdiv(3 * 16 * 512, 256), 256, 0, stream>>>(Wih, wih_p, RTD, 3, 256, 16);
  pack_weight_kernel<<<cdiv(2 * 16 * 512, 256), 256, 0, stream>>>(Whh, whh_p, HID, 2, 256, 16);
  pack_weight_kernel<<<cdiv(2 * 4 * 512, 256), 256, 0, stream>>>(Wn, wn_p, NFD, 2, HID, 4);
  pack_weight_kernel<<<cdiv(2 * 16 * 512, 256), 256, 0, stream>>>(W1, w1_p, HID, 2, 256, 16);
  pack_weight_kernel<<<cdiv(8 * 16 * 512, 256), 256, 0, stream>>>(W2, w2_p, 256, 8, 256, 16);

  // ---- fused LSTM + attention + static branch -> h0 (BN_TOTAL x 64)
  lstm_fused_kernel<<<BN_TOTAL / 32, 64, 0, stream>>>(x, wih_p, whh_p, wn_p, bih, bhh,
                                                      Wa, ba, bnb, h0);

  // ordered-uint encoding of -3e38 for segment-max init
  float neg = -3.0e38f;
  unsigned int nb;
  memcpy(&nb, &neg, 4);
  const unsigned int encneg = ~nb;   // negative float -> ~bits

  const int nEH   = BN_TOTAL * HEADS;           // 160000
  const int edgeT = BATCH * EE * HEADS;         // per-(b,e,h) threads
  const int aggW  = BATCH * EE;                 // one wave per (b,e)

  for (int layer = 0; layer < 2; ++layer) {
    const float* hin  = (layer == 0) ? h0 : hbn;
    const unsigned short* wp = (layer == 0) ? w1_p : w2_p;
    const float* a_s  = (layer == 0) ? as1 : as2;
    const float* a_d  = (layer == 0) ? ad1 : ad2;
    const float* bias = (layer == 0) ? b1 : b2;
    const float* gam  = (layer == 0) ? g1 : g2;
    const float* bet  = (layer == 0) ? be1 : be2;

    // init per-layer scratch
    fill_u32_kernel<<<cdiv(nEH, 256), 256, 0, stream>>>(emax, encneg, nEH);
    fill_f32_kernel<<<cdiv(nEH, 256), 256, 0, stream>>>(den, 0.f, nEH);
    fill_f32_kernel<<<cdiv(BN_TOTAL * CCH, 256), 256, 0, stream>>>(agg, 0.f, BN_TOTAL * CCH);
    fill_f32_kernel<<<cdiv(2 * CCH, 256), 256, 0, stream>>>(stat, 0.f, 2 * CCH);

    // xw = h @ W^T  (WMMA)
    if (layer == 0)
      gemm_xw_kernel<64><<<BN_TOTAL / 32, 64, 0, stream>>>(hin, wp, xw);
    else
      gemm_xw_kernel<256><<<BN_TOTAL / 32, 64, 0, stream>>>(hin, wp, xw);

    // per-node attention coefficients
    attn_coef_kernel<<<cdiv(nEH, 256), 256, 0, stream>>>(xw, a_s, a_d, asrc, adst, nEH);

    // segment softmax over edges
    edge_max_kernel<<<cdiv(edgeT, 256), 256, 0, stream>>>(ei, E, EE, asrc, adst, emax, edgeT);
    dec_max_kernel<<<cdiv(nEH, 256), 256, 0, stream>>>(emax, nEH);
    edge_sum_kernel<<<cdiv(edgeT, 256), 256, 0, stream>>>(ei, E, EE, asrc, adst,
                                                          (const float*)emax, den, edgeT);
    edge_agg_kernel<<<cdiv(aggW * 32, 256), 256, 0, stream>>>(ei, E, EE, asrc, adst,
                                                              (const float*)emax, den,
                                                              xw, agg, aggW);

    // bias + relu + BN
    bias_relu_stats_kernel<<<cdiv(BN_TOTAL, 64), 256, 0, stream>>>(agg, bias, stat, stat + CCH);
    bn_apply_kernel<<<cdiv(BN_TOTAL * CCH, 256), 256, 0, stream>>>(agg, stat, stat + CCH,
                                                                   gam, bet, hbn,
                                                                   BN_TOTAL * CCH);
  }

  // final projection -> (B, N)
  final_dot_kernel<<<cdiv(BN_TOTAL * 32, 256), 256, 0, stream>>>(hbn, Wc, bc, out);
}
